// Model_86938728006046
// MI455X (gfx1250) — matrix operand order repaired and checked by reading, not compiled, not run on hardware
//
#include <hip/hip_runtime.h>
#include <hip/hip_bf16.h>
#include <math.h>

// ---------------- CDNA5 WMMA types ----------------
typedef __attribute__((ext_vector_type(16))) __bf16 v16bf;
typedef __attribute__((ext_vector_type(8)))  float  v8f;

union AFragU { struct { uint4 lo, hi; } q; v16bf v; };

// ---------------- constants ----------------
#define N_ITEMS 50000
#define LOC_E 500
#define CID_E 100
#define UID_E 100
#define HID 500
#define LONG_E 650           // loc 500 + tim 50 + cid 100
#define SEQ_LONG 1024        // S - TGT
#define TGT 64

__device__ __forceinline__ float selu_f(float x) {
    const float sc = 1.0507009873554805f, al = 1.6732632423543772f;
    return x > 0.f ? sc * x : sc * al * (expf(x) - 1.f);
}
__device__ __forceinline__ float sigm_f(float x) { return 1.f / (1.f + expf(-x)); }

__device__ __forceinline__ float wave_sum(float v) {
    for (int m = 16; m > 0; m >>= 1) v += __shfl_xor(v, m, 32);
    return v;
}
__device__ __forceinline__ float wave_max(float v) {
    for (int m = 16; m > 0; m >>= 1) v = fmaxf(v, __shfl_xor(v, m, 32));
    return v;
}
__device__ float block_sum(float v, float* sd) {
    int tid = threadIdx.x, nw = blockDim.x >> 5;
    v = wave_sum(v);
    if ((tid & 31) == 0) sd[tid >> 5] = v;
    __syncthreads();
    float r = (tid < nw) ? sd[tid] : 0.f;
    if (tid < 32) { r = wave_sum(r); if (tid == 0) sd[0] = r; }
    __syncthreads();
    r = sd[0];
    __syncthreads();
    return r;
}
__device__ float block_max(float v, float* sd) {
    int tid = threadIdx.x, nw = blockDim.x >> 5;
    v = wave_max(v);
    if ((tid & 31) == 0) sd[tid >> 5] = v;
    __syncthreads();
    float r = (tid < nw) ? sd[tid] : -INFINITY;
    if (tid < 32) { r = wave_max(r); if (tid == 0) sd[0] = r; }
    __syncthreads();
    r = sd[0];
    __syncthreads();
    return r;
}

// ---------------- stage 1: uids_emb = Wl @ uvec + bl  [650] ----------------
__global__ void k_ue(const float* __restrict__ uid_table, const int* __restrict__ uid,
                     const float* __restrict__ Wl, const float* __restrict__ bl,
                     float* __restrict__ ue) {
    int r = blockIdx.x * blockDim.x + threadIdx.x;
    if (r >= LONG_E) return;
    const float* uvec = uid_table + (size_t)uid[0] * UID_E;
    const float* w = Wl + (size_t)r * UID_E;
    float acc = bl[r];
    for (int k = 0; k < UID_E; k++) acc += w[k] * uvec[k];
    ue[r] = acc;
}

// ---------------- stage 2: logits[i] = long_emb[i] . ue ----------------
__global__ void k_logit(const int* __restrict__ loc, const int* __restrict__ tim,
                        const int* __restrict__ cid,
                        const float* __restrict__ loc_t, const float* __restrict__ tim_t,
                        const float* __restrict__ cid_t,
                        const float* __restrict__ ue, float* __restrict__ logit) {
    __shared__ float sd[8];
    int i = blockIdx.x, tid = threadIdx.x;
    int l = loc[i], t = tim[i], c = cid[i];
    float acc = 0.f;
    for (int j = tid; j < LONG_E; j += blockDim.x) {
        float e = (j < 500) ? loc_t[(size_t)l * 500 + j]
                : (j < 550) ? tim_t[(size_t)t * 50 + (j - 500)]
                            : cid_t[(size_t)c * 100 + (j - 550)];
        acc += e * ue[j];
    }
    acc = block_sum(acc, sd);
    if (tid == 0) logit[i] = acc;
}

// ---------------- stage 3: attn = softmax(logits) over 1024 ----------------
__global__ void k_attn(const float* __restrict__ logit, float* __restrict__ attn) {
    __shared__ float sd[32];
    int tid = threadIdx.x;
    float v = logit[tid];
    float m = block_max(v, sd);
    float e = expf(v - m);
    float s = block_sum(e, sd);
    attn[tid] = e / s;
}

// ---------------- stage 4: long_res[j] = sum_i attn[i] * emb(i,j) ----------------
__global__ void k_longres(const float* __restrict__ attn,
                          const int* __restrict__ loc, const int* __restrict__ tim,
                          const int* __restrict__ cid,
                          const float* __restrict__ loc_t, const float* __restrict__ tim_t,
                          const float* __restrict__ cid_t, float* __restrict__ lres) {
    int j = blockIdx.x * blockDim.x + threadIdx.x;
    if (j >= LONG_E) return;
    float acc = 0.f;
    for (int i = 0; i < SEQ_LONG; i++) {
        float e;
        if (j < 500)      e = loc_t[(size_t)loc[i] * 500 + j];
        else if (j < 550) e = tim_t[(size_t)tim[i] * 50 + (j - 500)];
        else              e = cid_t[(size_t)cid[i] * 100 + (j - 550)];
        acc += attn[i] * e;
    }
    lres[j] = acc;
}

// ---------------- stage 5: lp = selu(Wp @ long_res + bp)  (mat-vec, BW-bound) ----
__global__ void k_lp(const float* __restrict__ Wp, const float* __restrict__ bp,
                     const float* __restrict__ lres, float* __restrict__ lp) {
    int gw = (blockIdx.x * blockDim.x + threadIdx.x) >> 5;
    int lane = threadIdx.x & 31;
    if (gw >= N_ITEMS) return;
    const float* wrow = Wp + (size_t)gw * LONG_E;
    float acc = 0.f;
    for (int j = lane; j < LONG_E; j += 32) acc += wrow[j] * lres[j];
    acc = wave_sum(acc);
    if (lane == 0) lp[gw] = selu_f(acc + bp[gw]);
}

// ---------------- stage 6: build bf16 X matrices (K-padded) + zero H pads ------
__global__ void k_buildX(const float* __restrict__ uid_table, const int* __restrict__ uid,
                         const float* __restrict__ tim_t, const float* __restrict__ loc_t,
                         const float* __restrict__ cid_t,
                         const int* __restrict__ loc, const int* __restrict__ tim,
                         const int* __restrict__ cid,
                         __bf16* __restrict__ X1, __bf16* __restrict__ X2,
                         __bf16* __restrict__ H1, __bf16* __restrict__ H2) {
    int t = blockIdx.x, tid = threadIdx.x;
    const float* uvec = uid_table + (size_t)uid[0] * UID_E;
    int ll = loc[SEQ_LONG + t], tt = tim[SEQ_LONG + t], cc = cid[SEQ_LONG + t];
    for (int j = tid; j < 672; j += blockDim.x) {            // x_loc: [uid100, tim50, loc500] pad->672
        float v = (j < 100) ? uvec[j]
                : (j < 150) ? tim_t[(size_t)tt * 50 + (j - 100)]
                : (j < 650) ? loc_t[(size_t)ll * 500 + (j - 150)] : 0.f;
        X1[(size_t)t * 672 + j] = (__bf16)v;
    }
    for (int j = tid; j < 256; j += blockDim.x) {            // x_cid: [uid100, tim50, cid100] pad->256
        float v = (j < 100) ? uvec[j]
                : (j < 150) ? tim_t[(size_t)tt * 50 + (j - 100)]
                : (j < 250) ? cid_t[(size_t)cc * 100 + (j - 150)] : 0.f;
        X2[(size_t)t * 256 + j] = (__bf16)v;
    }
    for (int j = tid; j < 512; j += blockDim.x) {            // zero K-pad region of H (cols 500..511)
        H1[(size_t)t * 512 + j] = (__bf16)0.f;
        H2[(size_t)t * 512 + j] = (__bf16)0.f;
    }
}

// ---------------- stage 7: Whh -> transposed bf16 (coalesced recurrent reads) --
__global__ void k_transw(const float* __restrict__ Whh1, const float* __restrict__ Whh2,
                         __bf16* __restrict__ WT1, __bf16* __restrict__ WT2) {
    int idx = blockIdx.x * blockDim.x + threadIdx.x;
    const int per = 2000 * 500;
    if (idx >= 2 * per) return;
    int which = idx / per, rem = idx % per;
    int k = rem / 2000, r = rem % 2000;                      // write WT[k][r]: coalesced in r
    const float* W = which ? Whh2 : Whh1;
    __bf16* WT = which ? WT2 : WT1;
    WT[(size_t)k * 2000 + r] = (__bf16)W[(size_t)r * 500 + k];
}

// ---------------- WMMA GEMM: out[64,N] = A[64,KPAD]bf16 x W[N,KACT]^T + bias ---
// Fully compile-time shape: TWO selects 32-col waves (two 16-col tiles sharing
// each A fragment -> half the A-side traffic) vs 16-col remainder waves.
// Straight-line inner loop: no divergent branches around WMMA, immediate-offset
// A loads, peeled clamped K-tail (no EXEC churn). W fp32 read once, cvt in-reg.
template <int KPAD, int KACT, bool TWO>
__global__ void k_gemm64(const __bf16* __restrict__ A,
                         const float* __restrict__ W, int N, int n_base,
                         const float* __restrict__ bias,
                         float* __restrict__ out, int do_selu) {
    constexpr int KFULL = KACT / 32;                         // full 32-wide k-tiles
    constexpr bool HAS_TAIL = (KPAD / 32) > KFULL;
    int lane = threadIdx.x & 31;
    int wave = threadIdx.x >> 5;
    int n0 = n_base + blockIdx.x * (TWO ? 128 : 64) + wave * (TWO ? 32 : 16);
    if (n0 >= N) return;                                     // wave-uniform, no barriers
    int nlo = lane & 15;
    int grp = lane >> 4;                                     // 0 or 1
    int nA = n0 + nlo;
    int nB = n0 + 16 + nlo;                                  // valid by construction if TWO
    const float* wrowA = W + (size_t)nA * KACT;
    const float* wrowB = TWO ? (W + (size_t)nB * KACT) : wrowA;

    v8f acc0[4], acc1[4];
#pragma unroll
    for (int mt = 0; mt < 4; mt++)
#pragma unroll
        for (int i = 0; i < 8; i++) { acc0[mt][i] = 0.f; acc1[mt][i] = 0.f; }

    const __bf16* abase = A + (size_t)nlo * KPAD + grp * 16; // + kt*32 per iter
    const float*  wbase = (const float*)nullptr;

    for (int kt = 0; kt < KFULL; kt++) {
        // B fragments: 16 contiguous K values of the W row (32x16 bf16 layout)
        const float* wpA = wrowA + kt * 32 + grp * 16;
        v16bf bfA, bfB;
#pragma unroll
        for (int j = 0; j < 8; j++) {
            float2 w2 = ((const float2*)wpA)[j];
            bfA[2 * j] = (__bf16)w2.x; bfA[2 * j + 1] = (__bf16)w2.y;
        }
        if (TWO) {
            const float* wpB = wrowB + kt * 32 + grp * 16;
#pragma unroll
            for (int j = 0; j < 8; j++) {
                float2 w2 = ((const float2*)wpB)[j];
                bfB[2 * j] = (__bf16)w2.x; bfB[2 * j + 1] = (__bf16)w2.y;
            }
        }
        // A fragments: per-lane k = {b..b+15}, b = grp*16 (+kt*32)
        const __bf16* ap = abase + kt * 32;
#pragma unroll
        for (int mt = 0; mt < 4; mt++) {                     // imm-offset loads: mt*16*KPAD
            AFragU af;
            af.q.lo = *(const uint4*)(ap + mt * 16 * KPAD);
            af.q.hi = *(const uint4*)(ap + mt * 16 * KPAD + 8);
            acc0[mt] = __builtin_amdgcn_wmma_f32_16x16x32_bf16(
                false, af.v, false, bfA, (short)0, acc0[mt], false, false);
            if (TWO)
                acc1[mt] = __builtin_amdgcn_wmma_f32_16x16x32_bf16(
                    false, af.v, false, bfB, (short)0, acc1[mt], false, false);
        }
    }
    if (HAS_TAIL) {                                          // clamped loads + select: no exec churn
        int kb = KFULL * 32 + grp * 16;
        v16bf bfA, bfB;
#pragma unroll
        for (int j = 0; j < 16; j++) {
            int k = kb + j;
            int kc = k < KACT ? k : KACT - 1;
            float va = wrowA[kc];
            bfA[j] = (__bf16)(k < KACT ? va : 0.f);
            if (TWO) {
                float vb = wrowB[kc];
                bfB[j] = (__bf16)(k < KACT ? vb : 0.f);
            }
        }
        const __bf16* ap = abase + KFULL * 32;
#pragma unroll
        for (int mt = 0; mt < 4; mt++) {
            AFragU af;
            af.q.lo = *(const uint4*)(ap + mt * 16 * KPAD);
            af.q.hi = *(const uint4*)(ap + mt * 16 * KPAD + 8);
            acc0[mt] = __builtin_amdgcn_wmma_f32_16x16x32_bf16(
                false, af.v, false, bfA, (short)0, acc0[mt], false, false);
            if (TWO)
                acc1[mt] = __builtin_amdgcn_wmma_f32_16x16x32_bf16(
                    false, af.v, false, bfB, (short)0, acc1[mt], false, false);
        }
    }
    float bvA = bias[nA];
    float bvB = TWO ? bias[nB] : 0.f;
#pragma unroll
    for (int mt = 0; mt < 4; mt++) {
#pragma unroll
        for (int v = 0; v < 8; v++) {                        // D: M = grp*8+v, N = lane&15
            int row = mt * 16 + grp * 8 + v;
            float v0 = acc0[mt][v] + bvA;
            if (do_selu) v0 = selu_f(v0);
            out[(size_t)row * N + nA] = v0;
            if (TWO) {
                float v1 = acc1[mt][v] + bvB;
                if (do_selu) v1 = selu_f(v1);
                out[(size_t)row * N + nB] = v1;
            }
        }
    }
}

// helper: launch full-width (two-tile) region + 16-col remainder region
template <int KPAD, int KACT>
static void launch_gemm64(const __bf16* A, const float* W, int N,
                          const float* bias, float* out, int do_selu,
                          hipStream_t stream) {
    int full_blocks = N / 128;                               // each covers 128 cols, all valid
    int n_full = full_blocks * 128;
    if (full_blocks > 0)
        k_gemm64<KPAD, KACT, true><<<full_blocks, 128, 0, stream>>>(
            A, W, N, 0, bias, out, do_selu);
    int rem = N - n_full;                                    // multiple of 16
    if (rem > 0)
        k_gemm64<KPAD, KACT, false><<<(rem + 63) / 64, 128, 0, stream>>>(
            A, W, N, n_full, bias, out, do_selu);
}

// ---------------- stage 9: two LSTMs (grid.x selects), recurrent part only -----
__global__ void k_lstm(const float* __restrict__ XP1, const float* __restrict__ XP2,
                       const __bf16* __restrict__ WT1, const __bf16* __restrict__ WT2,
                       __bf16* __restrict__ H1, __bf16* __restrict__ H2) {
    const float*  XP = blockIdx.x == 0 ? XP1 : XP2;
    const __bf16* WT = blockIdx.x == 0 ? WT1 : WT2;
    __bf16*       H  = blockIdx.x == 0 ? H1 : H2;
    __shared__ float hs[HID], cs[HID], zs[4 * HID];
    int tid = threadIdx.x;
    if (tid < HID) { hs[tid] = 0.f; cs[tid] = 0.f; }
    __syncthreads();
    int r0 = tid, r1 = tid + 1024;
    bool has2 = (r1 < 4 * HID);
    for (int t = 0; t < TGT; t++) {
        float a0 = 0.f, a1 = 0.f;
        for (int k = 0; k < HID; k++) {
            float hk = hs[k];
            const __bf16* wk = WT + (size_t)k * 2000;
            a0 += (float)wk[r0] * hk;
            if (has2) a1 += (float)wk[r1] * hk;
        }
        zs[r0] = XP[(size_t)t * 2000 + r0] + a0;
        if (has2) zs[r1] = XP[(size_t)t * 2000 + r1] + a1;
        __syncthreads();
        if (tid < HID) {                                     // gate order i,f,g,o
            float ig = sigm_f(zs[tid]);
            float fg = sigm_f(zs[HID + tid]);
            float gg = tanhf(zs[2 * HID + tid]);
            float og = sigm_f(zs[3 * HID + tid]);
            float c = fg * cs[tid] + ig * gg;
            float h = og * tanhf(c);
            cs[tid] = c; hs[tid] = h;
            H[(size_t)t * 512 + tid] = (__bf16)h;
        }
        __syncthreads();
    }
}

// ---------------- stage 11: score = log_softmax(lp + sl + sc) per row ----------
// (inner log_softmax constants cancel in the outer one -> skip them entirely)
__global__ void k_combine(const float* __restrict__ lp, float* __restrict__ slr,
                          const float* __restrict__ scr, float* __restrict__ out) {
    __shared__ float sd[32];
    int row = blockIdx.x, tid = threadIdx.x;
    size_t roff = (size_t)row * N_ITEMS;
    float m = -INFINITY;
    for (int n = tid; n < N_ITEMS; n += blockDim.x) {
        float g = lp[n] + slr[roff + n] + scr[roff + n];
        slr[roff + n] = g;                                   // in-place: same index, safe
        m = fmaxf(m, g);
    }
    m = block_max(m, sd);
    float s = 0.f;
    for (int n = tid; n < N_ITEMS; n += blockDim.x) s += expf(slr[roff + n] - m);
    s = block_sum(s, sd);
    float ls = logf(s);
    for (int n = tid; n < N_ITEMS; n += blockDim.x)
        out[roff + n] = slr[roff + n] - m - ls;
}

// =====================================================================
extern "C" void kernel_launch(void* const* d_in, const int* in_sizes, int n_in,
                              void* d_out, int out_size, void* d_ws, size_t ws_size,
                              hipStream_t stream) {
    (void)in_sizes; (void)n_in; (void)out_size; (void)ws_size;
    const int*   loc   = (const int*)d_in[0];
    const int*   tim   = (const int*)d_in[1];
    const int*   cid   = (const int*)d_in[2];
    const int*   uid   = (const int*)d_in[3];
    const float* loc_t = (const float*)d_in[6];
    const float* tim_t = (const float*)d_in[7];
    const float* uid_t = (const float*)d_in[8];
    const float* cid_t = (const float*)d_in[9];
    const float* Wl    = (const float*)d_in[10];
    const float* bl    = (const float*)d_in[11];
    const float* Wp    = (const float*)d_in[12];
    const float* bp    = (const float*)d_in[13];
    const float* Wsl   = (const float*)d_in[14];
    const float* bsl   = (const float*)d_in[15];
    const float* Wsc   = (const float*)d_in[16];
    const float* bsc   = (const float*)d_in[17];
    const float* Wih1  = (const float*)d_in[18];
    const float* Whh1  = (const float*)d_in[19];
    const float* b1    = (const float*)d_in[20];
    const float* Wih2  = (const float*)d_in[21];
    const float* Whh2  = (const float*)d_in[22];
    const float* b2    = (const float*)d_in[23];
    float* out = (float*)d_out;

    // workspace carve-up (256B aligned)
    char* base = (char*)d_ws;
    size_t off = 0;
    auto alloc = [&](size_t bytes) -> char* {
        off = (off + 255) & ~(size_t)255;
        char* p = base + off;
        off += bytes;
        return p;
    };
    float*  ue   = (float*)alloc(LONG_E * 4);
    float*  lg   = (float*)alloc(SEQ_LONG * 4);
    float*  attn = (float*)alloc(SEQ_LONG * 4);
    float*  lres = (float*)alloc(LONG_E * 4);
    float*  lp   = (float*)alloc((size_t)N_ITEMS * 4);
    __bf16* X1   = (__bf16*)alloc((size_t)TGT * 672 * 2);
    __bf16* X2   = (__bf16*)alloc((size_t)TGT * 256 * 2);
    float*  XP1  = (float*)alloc((size_t)TGT * 2000 * 4);
    float*  XP2  = (float*)alloc((size_t)TGT * 2000 * 4);
    __bf16* WT1  = (__bf16*)alloc((size_t)HID * 2000 * 2);
    __bf16* WT2  = (__bf16*)alloc((size_t)HID * 2000 * 2);
    __bf16* H1   = (__bf16*)alloc((size_t)TGT * 512 * 2);
    __bf16* H2   = (__bf16*)alloc((size_t)TGT * 512 * 2);
    float*  slr  = (float*)alloc((size_t)TGT * N_ITEMS * 4);
    float*  scr  = (float*)alloc((size_t)TGT * N_ITEMS * 4);

    // long-term branch
    k_ue     <<<3, 256, 0, stream>>>(uid_t, uid, Wl, bl, ue);
    k_logit  <<<SEQ_LONG, 256, 0, stream>>>(loc, tim, cid, loc_t, tim_t, cid_t, ue, lg);
    k_attn   <<<1, 1024, 0, stream>>>(lg, attn);
    k_longres<<<3, 256, 0, stream>>>(attn, loc, tim, cid, loc_t, tim_t, cid_t, lres);
    k_lp     <<<6250, 256, 0, stream>>>(Wp, bp, lres, lp);   // 8 waves/block, 1 row/wave

    // short-term branch prep
    k_buildX <<<TGT, 256, 0, stream>>>(uid_t, uid, tim_t, loc_t, cid_t, loc, tim, cid,
                                       X1, X2, H1, H2);
    k_transw <<<(2 * 2000 * 500 + 255) / 256, 256, 0, stream>>>(Whh1, Whh2, WT1, WT2);

    // batched input projections (WMMA): XP = X @ Wih^T + b
    launch_gemm64<672, 650>(X1, Wih1, 2000, b1, XP1, 0, stream);
    launch_gemm64<256, 250>(X2, Wih2, 2000, b2, XP2, 0, stream);

    // sequential recurrence (both LSTMs in parallel)
    k_lstm   <<<2, 1024, 0, stream>>>(XP1, XP2, WT1, WT2, H1, H2);

    // score GEMMs (WMMA): sl = selu(H @ Wsl^T + bsl), sc likewise
    launch_gemm64<512, 500>(H1, Wsl, N_ITEMS, bsl, slr, 1, stream);
    launch_gemm64<512, 500>(H2, Wsc, N_ITEMS, bsc, scr, 1, stream);

    // final fused log-softmax
    k_combine<<<TGT, 1024, 0, stream>>>(lp, slr, scr, out);
}
